// GCNEntPair_71159018160437
// MI455X (gfx1250) — matrix operand order repaired and checked by reading, not compiled, not run on hardware
//
#include <hip/hip_runtime.h>

// ---------------------------------------------------------------------------
// GCN-Entity-Pair pipeline for MI455X (gfx1250, wave32, WMMA bf16).
// Dense layers -> v_wmma_f32_16x16x32_bf16 (f32 accumulate) with 32x64
// register blocking (8 WMMA per k-step, 1.5 b128-loads/WMMA); scatter ->
// f32 atomics hitting L2 (h arrays are 51MB << 192MB L2).
// ---------------------------------------------------------------------------

typedef __bf16 bf16_t;
typedef __attribute__((ext_vector_type(16))) __bf16 v16bf;
typedef __attribute__((ext_vector_type(8)))  __bf16 v8bf;
typedef __attribute__((ext_vector_type(8)))  float  v8f;

#define NN     50000
#define EE     800000
#define GG     1024
#define D_EMB  128
#define D_GH   256
#define D_EH   256
#define D_OUT  128
#define D_HID  512

// ----------------------------- elementwise kernels -------------------------

__global__ void gcnep_fill_f32(float* __restrict__ p, float v, int n) {
  int i = blockIdx.x * blockDim.x + threadIdx.x;
  if (i < n) p[i] = v;
}

// W[K,Nout] f32 -> Wt[Nout,K] bf16 (transposed for contiguous B-fragment loads)
__global__ void gcnep_wt_bf16(const float* __restrict__ W, bf16_t* __restrict__ Wt,
                              int K, int Nout) {
  int i = blockIdx.x * blockDim.x + threadIdx.x;
  if (i >= K * Nout) return;
  int k = i / Nout, n = i % Nout;
  Wt[(size_t)n * K + k] = (bf16_t)W[i];
}

// out[r,:] = (relu?) emb[idx[r],:]  as bf16
__global__ void gcnep_gather_emb(const float* __restrict__ emb, const int* __restrict__ idx,
                                 bf16_t* __restrict__ out, int rows, int dim, int do_relu) {
  int i = blockIdx.x * blockDim.x + threadIdx.x;
  if (i >= rows * dim) return;
  int r = i / dim, j = i % dim;
  float v = emb[(size_t)idx[r] * dim + j];
  if (do_relu) v = fmaxf(v, 0.0f);
  out[i] = (bf16_t)v;
}

__global__ void gcnep_deg(const int* __restrict__ dst, float* __restrict__ deg, int e) {
  int i = blockIdx.x * blockDim.x + threadIdx.x;
  if (i < e) atomicAdd(&deg[dst[i]], 1.0f);
}

// dinv = rsqrt(in_degree + 1 self loop)
__global__ void gcnep_dinv(const float* __restrict__ deg, float* __restrict__ dinv, int n) {
  int i = blockIdx.x * blockDim.x + threadIdx.x;
  if (i < n) dinv[i] = rsqrtf(deg[i] + 1.0f);
}

// agg[dst] += dinv[src]*dinv[dst]*h[src]  (dim fixed to 256, 4 floats/thread)
__global__ void gcnep_scatter(const int* __restrict__ src, const int* __restrict__ dst,
                              const float* __restrict__ dinv, const float* __restrict__ h,
                              float* __restrict__ agg, int e) {
  int t = blockIdx.x * blockDim.x + threadIdx.x;
  if (t >= e * 64) return;
  int ei = t >> 6;
  int j  = (t & 63) << 2;
  int s = src[ei], d = dst[ei];
  float w = dinv[s] * dinv[d];
  float4 hv = *(const float4*)(h + (size_t)s * D_GH + j);
  float* ap = agg + (size_t)d * D_GH + j;
  atomicAdd(ap + 0, w * hv.x);
  atomicAdd(ap + 1, w * hv.y);
  atomicAdd(ap + 2, w * hv.z);
  atomicAdd(ap + 3, w * hv.w);
}

// act = relu(agg + dinv^2 * h + bias)  -> bf16 (self loop folded in)
__global__ void gcnep_conv_post(const float* __restrict__ agg, const float* __restrict__ h,
                                const float* __restrict__ dinv, const float* __restrict__ bias,
                                bf16_t* __restrict__ act, int n, int dim) {
  int i = blockIdx.x * blockDim.x + threadIdx.x;
  if (i >= n * dim) return;
  int r = i / dim, j = i % dim;
  float di = dinv[r];
  float v = agg[i] + di * di * h[i] + bias[j];
  act[i] = (bf16_t)fmaxf(v, 0.0f);
}

__global__ void gcnep_pool_cnt(const int* __restrict__ batch, float* __restrict__ cnt, int n) {
  int i = blockIdx.x * blockDim.x + threadIdx.x;
  if (i < n) atomicAdd(&cnt[batch[i]], 1.0f);
}

__global__ void gcnep_pool_sum(const int* __restrict__ batch, const bf16_t* __restrict__ act,
                               float* __restrict__ psum, int n, int dim) {
  int i = blockIdx.x * blockDim.x + threadIdx.x;
  if (i >= n * dim) return;
  int r = i / dim, j = i % dim;
  atomicAdd(&psum[(size_t)batch[r] * dim + j], (float)act[i]);
}

__global__ void gcnep_pool_fin(const float* __restrict__ psum, const float* __restrict__ cnt,
                               bf16_t* __restrict__ out, int g, int dim) {
  int i = blockIdx.x * blockDim.x + threadIdx.x;
  if (i >= g * dim) return;
  int r = i / dim;
  out[i] = (bf16_t)(psum[i] / fmaxf(cnt[r], 1.0f));
}

// egs = relu(concat(g1,e1) + concat(g2,e2)) -> bf16 [G, HID]
__global__ void gcnep_combine(const float* __restrict__ g1, const float* __restrict__ e1,
                              const float* __restrict__ g2, const float* __restrict__ e2,
                              bf16_t* __restrict__ egs, int g, int gh, int eh) {
  int hid = gh + eh;
  int i = blockIdx.x * blockDim.x + threadIdx.x;
  if (i >= g * hid) return;
  int r = i / hid, j = i % hid;
  float a, b;
  if (j < gh) { a = g1[(size_t)r * gh + j];        b = g2[(size_t)r * gh + j]; }
  else        { a = e1[(size_t)r * eh + (j - gh)]; b = e2[(size_t)r * eh + (j - gh)]; }
  egs[i] = (bf16_t)fmaxf(a + b, 0.0f);
}

// ----------------------------- WMMA GEMM -----------------------------------
// C[M,Nc] = A[M,K](bf16, row-major) @ Bt[Nc,K](bf16, pre-transposed) (+bias)(+relu)
// One wave computes a 32x64 strip (2 M-tiles x 4 N-tiles) of C: 8 independent
// v_wmma_f32_16x16x32_bf16 per k-step (no D->A/B hazards), 12 b128 loads per
// 8 WMMAs. Requires: Nc%64==0, K%32==0, 16B-aligned buffers. M remainder
// handled by clamping A row indices (EXEC stays all-ones across the WMMA,
// ISA 7.12 requirement) and guarding stores.
__global__ void gcnep_gemm_wmma(const bf16_t* __restrict__ A, const bf16_t* __restrict__ Bt,
                                const float* __restrict__ bias,
                                float* __restrict__ Cf, bf16_t* __restrict__ Cb,
                                int M, int Nc, int K, int do_relu) {
  const int lane   = threadIdx.x & 31;
  const int wave   = threadIdx.x >> 5;
  const int tilesN = Nc >> 6;                  // 64-wide N strips
  const int tilesM = (M + 31) >> 5;            // 32-tall M strips
  const int nTiles = tilesM * tilesN;
  const int tile = blockIdx.x * (blockDim.x >> 5) + wave;
  if (tile >= nTiles) return;                  // uniform per wave
  const int tm   = tile / tilesN;
  const int tn   = tile % tilesN;
  const int half = lane >> 4;                  // 0: lanes 0-15, 1: lanes 16-31
  const int l16  = lane & 15;

  // A rows for the two M-tiles (clamped so partial last strip stays in-bounds)
  int r0 = tm * 32 + l16;      if (r0 > M - 1) r0 = M - 1;
  int r1 = tm * 32 + 16 + l16; if (r1 > M - 1) r1 = M - 1;
  const bf16_t* arow0 = A + (size_t)r0 * K;
  const bf16_t* arow1 = A + (size_t)r1 * K;
  const bf16_t* brow0 = Bt + (size_t)(tn * 64 +  0 + l16) * K;
  const bf16_t* brow1 = Bt + (size_t)(tn * 64 + 16 + l16) * K;
  const bf16_t* brow2 = Bt + (size_t)(tn * 64 + 32 + l16) * K;
  const bf16_t* brow3 = Bt + (size_t)(tn * 64 + 48 + l16) * K;

  v8f acc[2][4] = {};
  for (int k0 = 0; k0 < K; k0 += 32) {
    // A 16x32 fragment: elems 0-7 = K[k0+8h .. +7], elems 8-15 = K[k0+16+8h .. +7]
    v8bf alo0 = *(const v8bf*)(arow0 + k0 + half * 8);
    v8bf ahi0 = *(const v8bf*)(arow0 + k0 + 16 + half * 8);
    v8bf alo1 = *(const v8bf*)(arow1 + k0 + half * 8);
    v8bf ahi1 = *(const v8bf*)(arow1 + k0 + 16 + half * 8);
    // B 32x16 fragment: lane=col, elems 0-7 = K[k0+8h .. +7], elems 8-15 = K[k0+16+8h .. +7]
    const bf16_t* brows[4] = { brow0, brow1, brow2, brow3 };
    v16bf a[2], b[4];
#pragma unroll
    for (int t = 0; t < 8; ++t) {
      a[0][t] = alo0[t]; a[0][t + 8] = ahi0[t];
      a[1][t] = alo1[t]; a[1][t + 8] = ahi1[t];
    }
#pragma unroll
    for (int j = 0; j < 4; ++j) {
      v8bf blo = *(const v8bf*)(brows[j] + k0 + half * 8);
      v8bf bhi = *(const v8bf*)(brows[j] + k0 + 16 + half * 8);
#pragma unroll
      for (int t = 0; t < 8; ++t) { b[j][t] = blo[t]; b[j][t + 8] = bhi[t]; }
    }
#pragma unroll
    for (int i = 0; i < 2; ++i)
#pragma unroll
      for (int j = 0; j < 4; ++j)
        acc[i][j] = __builtin_amdgcn_wmma_f32_16x16x32_bf16(
            /*neg_a=*/false, a[i], /*neg_b=*/false, b[j],
            /*c_mod=*/(short)0, acc[i][j], /*reuse_a=*/false, /*reuse_b=*/false);
  }

#pragma unroll
  for (int i = 0; i < 2; ++i) {
#pragma unroll
    for (int j = 0; j < 4; ++j) {
      const int col = tn * 64 + j * 16 + l16;
      const float bv = bias ? bias[col] : 0.0f;
#pragma unroll
      for (int v = 0; v < 8; ++v) {            // C/D: elem v = row v + 8*half
        int row = tm * 32 + i * 16 + v + half * 8;
        if (row < M) {
          float x = acc[i][j][v] + bv;
          if (do_relu) x = fmaxf(x, 0.0f);
          size_t o = (size_t)row * Nc + col;
          if (Cf) Cf[o] = x;
          if (Cb) Cb[o] = (bf16_t)x;
        }
      }
    }
  }
}

// ----------------------------- host orchestration --------------------------

extern "C" void kernel_launch(void* const* d_in, const int* in_sizes, int n_in,
                              void* d_out, int out_size, void* d_ws, size_t ws_size,
                              hipStream_t stream) {
  (void)in_sizes; (void)n_in; (void)out_size; (void)ws_size;

  const int*   x1   = (const int*)d_in[0];
  const int*   ei1  = (const int*)d_in[1];
  const int*   ent1 = (const int*)d_in[2];
  const int*   b1   = (const int*)d_in[3];
  const int*   x2   = (const int*)d_in[4];
  const int*   ei2  = (const int*)d_in[5];
  const int*   ent2 = (const int*)d_in[6];
  const int*   b2   = (const int*)d_in[7];
  const float* atom_emb = (const float*)d_in[8];
  const float* gW1 = (const float*)d_in[9];  const float* gb1 = (const float*)d_in[10];
  const float* gW2 = (const float*)d_in[11]; const float* gb2 = (const float*)d_in[12];
  const float* fcW = (const float*)d_in[13]; const float* fcb = (const float*)d_in[14];
  const float* ent_emb = (const float*)d_in[15];
  const float* eW1 = (const float*)d_in[16]; const float* eb1 = (const float*)d_in[17];
  const float* eW2 = (const float*)d_in[18]; const float* eb2 = (const float*)d_in[19];
  const float* dW1 = (const float*)d_in[20]; const float* db1 = (const float*)d_in[21];
  const float* dW2 = (const float*)d_in[22]; const float* db2 = (const float*)d_in[23];
  const float* dW3 = (const float*)d_in[24]; const float* db3 = (const float*)d_in[25];
  float* out = (float*)d_out;

  // --- workspace carve-out (256B aligned) ---
  char* base = (char*)d_ws;
  size_t off = 0;
  auto alloc = [&](size_t bytes) -> void* {
    off = (off + 255) & ~(size_t)255;
    void* p = base + off;
    off += bytes;
    return p;
  };

  bf16_t* wt_g1 = (bf16_t*)alloc((size_t)D_GH * D_EMB * 2);
  bf16_t* wt_g2 = (bf16_t*)alloc((size_t)D_GH * D_GH  * 2);
  bf16_t* wt_fc = (bf16_t*)alloc((size_t)D_GH * D_GH  * 2);
  bf16_t* wt_e1 = (bf16_t*)alloc((size_t)D_EH * D_EMB * 2);
  bf16_t* wt_e2 = (bf16_t*)alloc((size_t)D_EH * D_EH  * 2);
  bf16_t* wt_d1 = (bf16_t*)alloc((size_t)D_HID * D_HID * 2);
  bf16_t* wt_d2 = (bf16_t*)alloc((size_t)D_HID * D_HID * 2);
  bf16_t* wt_d3 = (bf16_t*)alloc((size_t)D_OUT * D_HID * 2);

  bf16_t* h0      = (bf16_t*)alloc((size_t)NN * D_EMB * 2);   // gathered atom emb
  float*  gemmout = (float*) alloc((size_t)NN * D_GH * 4);    // h = x@W (f32, L2-resident)
  float*  agg     = (float*) alloc((size_t)NN * D_GH * 4);    // edge aggregation
  bf16_t* act     = (bf16_t*)alloc((size_t)NN * D_GH * 2);    // relu'd conv output
  float*  deg     = (float*) alloc((size_t)NN * 4);
  float*  dinv    = (float*) alloc((size_t)NN * 4);
  float*  cnt     = (float*) alloc((size_t)GG * 4);
  float*  psum    = (float*) alloc((size_t)GG * D_GH * 4);

  bf16_t* gpool1 = (bf16_t*)alloc((size_t)GG * D_GH * 2);
  bf16_t* gpool2 = (bf16_t*)alloc((size_t)GG * D_GH * 2);
  float*  gout1  = (float*) alloc((size_t)GG * D_GH * 4);
  float*  gout2  = (float*) alloc((size_t)GG * D_GH * 4);
  bf16_t* e0buf  = (bf16_t*)alloc((size_t)GG * D_EMB * 2);
  bf16_t* e1buf  = (bf16_t*)alloc((size_t)GG * D_EH * 2);
  float*  eout1  = (float*) alloc((size_t)GG * D_EH * 4);
  float*  eout2  = (float*) alloc((size_t)GG * D_EH * 4);
  bf16_t* egs    = (bf16_t*)alloc((size_t)GG * D_HID * 2);
  bf16_t* dh1    = (bf16_t*)alloc((size_t)GG * D_HID * 2);
  bf16_t* dh2    = (bf16_t*)alloc((size_t)GG * D_HID * 2);

  auto blocks = [](int n) { return (n + 255) / 256; };

  auto gemm = [&](const bf16_t* A, const bf16_t* Bt, const float* bias,
                  float* Cf, bf16_t* Cb, int M, int Nc, int K, int relu) {
    int nTiles = ((M + 31) / 32) * (Nc / 64);
    dim3 grd((nTiles + 3) / 4), blk(128);   // 4 waves/block, 32x64 strip/wave
    gcnep_gemm_wmma<<<grd, blk, 0, stream>>>(A, Bt, bias, Cf, Cb, M, Nc, K, relu);
  };

  // --- weight transpose+bf16 conversion (tiny, once per call) ---
  auto wconv = [&](const float* W, bf16_t* Wt, int K, int Nout) {
    gcnep_wt_bf16<<<blocks(K * Nout), 256, 0, stream>>>(W, Wt, K, Nout);
  };
  wconv(gW1, wt_g1, D_EMB, D_GH);  wconv(gW2, wt_g2, D_GH, D_GH);
  wconv(fcW, wt_fc, D_GH, D_GH);
  wconv(eW1, wt_e1, D_EMB, D_EH);  wconv(eW2, wt_e2, D_EH, D_EH);
  wconv(dW1, wt_d1, D_HID, D_HID); wconv(dW2, wt_d2, D_HID, D_HID);
  wconv(dW3, wt_d3, D_HID, D_OUT);

  // --- one GCN branch: emb -> conv1 -> conv2 -> mean-pool -> fc ---
  auto run_branch = [&](const int* x, const int* ei, const int* batch,
                        bf16_t* gpool, float* gout) {
    const int* src = ei;
    const int* dst = ei + EE;

    gcnep_fill_f32<<<blocks(NN), 256, 0, stream>>>(deg, 0.0f, NN);
    gcnep_deg<<<blocks(EE), 256, 0, stream>>>(dst, deg, EE);
    gcnep_dinv<<<blocks(NN), 256, 0, stream>>>(deg, dinv, NN);

    gcnep_gather_emb<<<blocks(NN * D_EMB), 256, 0, stream>>>(atom_emb, x, h0, NN, D_EMB, 0);

    // conv1: h = x@W1 ; agg = scatter ; act = relu(agg + dinv^2*h + b1)
    gemm(h0, wt_g1, nullptr, gemmout, nullptr, NN, D_GH, D_EMB, 0);
    gcnep_fill_f32<<<blocks(NN * D_GH), 256, 0, stream>>>(agg, 0.0f, NN * D_GH);
    gcnep_scatter<<<blocks(EE * 64), 256, 0, stream>>>(src, dst, dinv, gemmout, agg, EE);
    gcnep_conv_post<<<blocks(NN * D_GH), 256, 0, stream>>>(agg, gemmout, dinv, gb1, act, NN, D_GH);

    // conv2
    gemm(act, wt_g2, nullptr, gemmout, nullptr, NN, D_GH, D_GH, 0);
    gcnep_fill_f32<<<blocks(NN * D_GH), 256, 0, stream>>>(agg, 0.0f, NN * D_GH);
    gcnep_scatter<<<blocks(EE * 64), 256, 0, stream>>>(src, dst, dinv, gemmout, agg, EE);
    gcnep_conv_post<<<blocks(NN * D_GH), 256, 0, stream>>>(agg, gemmout, dinv, gb2, act, NN, D_GH);

    // global mean pool + fc
    gcnep_fill_f32<<<blocks(GG), 256, 0, stream>>>(cnt, 0.0f, GG);
    gcnep_fill_f32<<<blocks(GG * D_GH), 256, 0, stream>>>(psum, 0.0f, GG * D_GH);
    gcnep_pool_cnt<<<blocks(NN), 256, 0, stream>>>(batch, cnt, NN);
    gcnep_pool_sum<<<blocks(NN * D_GH), 256, 0, stream>>>(batch, act, psum, NN, D_GH);
    gcnep_pool_fin<<<blocks(GG * D_GH), 256, 0, stream>>>(psum, cnt, gpool, GG, D_GH);
    gemm(gpool, wt_fc, fcb, gout, nullptr, GG, D_GH, D_GH, 0);   // no relu on fc
  };

  // --- entity encoder: relu(emb) -> relu(@eW1+b) -> relu(@eW2+b) ---
  auto run_ent = [&](const int* ent, float* eout) {
    gcnep_gather_emb<<<blocks(GG * D_EMB), 256, 0, stream>>>(ent_emb, ent, e0buf, GG, D_EMB, 1);
    gemm(e0buf, wt_e1, eb1, nullptr, e1buf, GG, D_EH, D_EMB, 1);
    gemm(e1buf, wt_e2, eb2, eout, nullptr, GG, D_EH, D_EH, 1);
  };

  run_branch(x1, ei1, b1, gpool1, gout1);
  run_branch(x2, ei2, b2, gpool2, gout2);
  run_ent(ent1, eout1);
  run_ent(ent2, eout2);

  // --- combine + decoder MLP ---
  gcnep_combine<<<blocks(GG * D_HID), 256, 0, stream>>>(gout1, eout1, gout2, eout2,
                                                        egs, GG, D_GH, D_EH);
  gemm(egs, wt_d1, db1, nullptr, dh1, GG, D_HID, D_HID, 1);
  gemm(dh1, wt_d2, db2, nullptr, dh2, GG, D_HID, D_HID, 1);
  gemm(dh2, wt_d3, db3, out, nullptr, GG, D_OUT, D_HID, 0);
}
